// InvariantPolynomial_85864986182260
// MI455X (gfx1250) — matrix operand order repaired and checked by reading, not compiled, not run on hardware
//
#include <hip/hip_runtime.h>

typedef __attribute__((ext_vector_type(16))) __bf16 v16bf;
typedef __attribute__((ext_vector_type(8)))  float  v8f;

#define IS3  0.5773502691896258f
#define IS5  0.4472135954999579f
#define IS6  0.4082482904638631f
#define IS10 0.31622776601683794f
#define IS30 0.18257418583505536f
#define IS70 0.11952286093343936f
#define S3C  1.7320508075688772f

// ---- per-path tables for G build (15 CG paths of fctp1) ----
__device__ const int   PW1OFF[15] = {0,4096,5632,6656,8192,12288,13824,14848,16384,17408,18432,19968,20992,25088,26624};
__device__ const int   PMO[15]    = {64,24,16,24,64,24,16,24,16,16,24,16,64,24,16};
__device__ const int   PK[15]     = {1,3,5,3,1,3,5,3,5,5,3,5,1,3,5};
__device__ const int   PKK[15]    = {0,1,4,9,12,13,16,21,24,29,34,37,42,43,46};
__device__ const int   POUT[15]   = {0,136,208,136,0,64,208,136,288,208,136,288,0,64,208};
__device__ const int   PSZ[15]    = {64,72,80,72,64,72,80,72,80,80,72,80,64,72,80};
__device__ const float PA8[15]    = {0.57735027f,0.86602540f,1.11803399f,0.86602540f,0.57735027f,
                                     1.22474487f,1.11803399f,0.86602540f,1.58113883f,1.11803399f,
                                     0.86602540f,1.58113883f,0.57735027f,1.22474487f,1.11803399f};

__global__ void k_edge_prep(const float* __restrict__ pos, const int* __restrict__ z,
                            const int* __restrict__ esrc, const int* __restrict__ edst, int E,
                            float* __restrict__ sh, int* __restrict__ zzbuf,
                            float* __restrict__ node1, int* __restrict__ cnt) {
  int e = blockIdx.x * blockDim.x + threadIdx.x;
  if (e >= E) return;
  int s = esrc[e], d = edst[e];
  float x = pos[3*s]   - pos[3*d];
  float y = pos[3*s+1] - pos[3*d+1];
  float zc= pos[3*s+2] - pos[3*d+2];
  const float s3 = 1.7320508075688772f, s15 = 3.872983346207417f, s5h = 1.118033988749895f;
  float v[9];
  v[0]=1.f; v[1]=s3*x; v[2]=s3*y; v[3]=s3*zc;
  v[4]=s15*x*y; v[5]=s15*y*zc; v[6]=s5h*(2.f*zc*zc - x*x - y*y);
  v[7]=s15*x*zc; v[8]=0.5f*s15*(x*x - y*y);
#pragma unroll
  for (int i = 0; i < 9; ++i) { sh[(size_t)e*9+i] = v[i]; atomicAdd(&node1[(size_t)d*9+i], v[i]); }
  int zz = 8*z[s] + z[d];
  zzbuf[e] = zz;
  atomicAdd(&cnt[zz], 1);
}

__global__ void k_scan_tiles(const int* __restrict__ cnt, int E, int* __restrict__ boff,
                             int* __restrict__ tile_z, int* __restrict__ tile_s, int* __restrict__ ntiles) {
  __shared__ int scnt[64], soff[64], stoff[64];
  int t = threadIdx.x;
  scnt[t] = cnt[t];
  __syncthreads();
  if (t == 0) {
    int o = 0, to = 0;
    for (int i = 0; i < 64; ++i) { soff[i] = o; stoff[i] = to; o += scnt[i]; to += (scnt[i] + 15) >> 4; }
    *ntiles = to;
    boff[64] = o;
  }
  __syncthreads();
  boff[t] = soff[t];
  int nt = (scnt[t] + 15) >> 4;
  int tb = stoff[t];
  for (int i = 0; i < nt; ++i) { tile_z[tb + i] = t; tile_s[tb + i] = soff[t] + 16*i; }
}

__global__ void k_scatter(const int* __restrict__ zzbuf, const int* __restrict__ boff,
                          int* __restrict__ cursor, int* __restrict__ order, int E) {
  int e = blockIdx.x * blockDim.x + threadIdx.x;
  if (e >= E) return;
  int zz = zzbuf[e];
  int p = atomicAdd(&cursor[zz], 1);
  order[boff[zz] + p] = e;
}

__global__ void k_gfill(const float* __restrict__ w1, __bf16* __restrict__ G) {
  int tid = blockIdx.x * blockDim.x + threadIdx.x;
  if (tid >= 64 * 1104) return;
  int z = tid / 1104, r = tid % 1104;
  int p = 0;
  while (r >= PSZ[p]) { r -= PSZ[p]; ++p; }
  int K = PK[p];
  int w = r / K, k = r % K;
  float val = PA8[p] * w1[PW1OFF[p] + z * PMO[p] + w];
  int row = PKK[p] + k;
  int col = POUT[p] + w * K + k;
  G[((size_t)z * 64 + row) * 368 + col] = (__bf16)val;
}

// Maps logical K index -> position in the WMMA-A-fragment-swizzled row layout,
// so that each lane's 16-element fragment is contiguous in memory.
__device__ __forceinline__ int a_swz(int k) {
  int s  = k >> 5;          // 32-wide K step
  int kk = k & 31;
  int h  = (kk >> 3) & 1;   // half-wave that owns this k
  int ix = (kk & 7) | ((kk >> 4) << 3);
  return (s << 5) + (h << 4) + ix;
}

__global__ void k_abuild(const int* __restrict__ order, const int* __restrict__ esrc,
                         const float* __restrict__ sh, const float* __restrict__ node1,
                         __bf16* __restrict__ A, int E) {
  int s = blockIdx.x * blockDim.x + threadIdx.x;
  if (s >= E) return;
  int e = order[s];
  int src = esrc[e];
  float a[9], b[9];
#pragma unroll
  for (int i = 0; i < 9; ++i) a[i] = node1[(size_t)src*9+i] * IS3;
#pragma unroll
  for (int i = 0; i < 9; ++i) b[i] = sh[(size_t)e*9+i];
  float a0=a[0], ax=a[1], ay=a[2], az=a[3], A0=a[4],A1=a[5],A2=a[6],A3=a[7],A4=a[8];
  float b0=b[0], bx=b[1], by=b[2], bz=b[3], B0=b[4],B1=b[5],B2=b[6],B3=b[7],B4=b[8];
  float T[64];
#pragma unroll
  for (int i = 51; i < 64; ++i) T[i] = 0.f;
  T[0]=a0*b0;
  T[1]=a0*bx*IS3; T[2]=a0*by*IS3; T[3]=a0*bz*IS3;
  T[4]=a0*B0*IS5; T[5]=a0*B1*IS5; T[6]=a0*B2*IS5; T[7]=a0*B3*IS5; T[8]=a0*B4*IS5;
  T[9]=ax*b0*IS3; T[10]=ay*b0*IS3; T[11]=az*b0*IS3;
  T[12]=(ax*bx+ay*by+az*bz)*IS3;
  T[13]=(ay*bz-az*by)*IS6; T[14]=(az*bx-ax*bz)*IS6; T[15]=(ax*by-ay*bx)*IS6;
  T[16]=(ax*by+ay*bx)*IS10; T[17]=(ay*bz+az*by)*IS10;
  T[18]=(2.f*az*bz-ax*bx-ay*by)*IS30;
  T[19]=(ax*bz+az*bx)*IS10; T[20]=(ax*bx-ay*by)*IS10;
  T[21]=(ay*B0+az*B3+ax*B4)*IS10 - ax*B2*IS30;
  T[22]=(ax*B0+az*B1-ay*B4)*IS10 - ay*B2*IS30;
  T[23]=(ay*B1+ax*B3)*IS10 + 2.f*az*B2*IS30;
  T[24]=(ax*B3 - ay*B1 - 2.f*az*B4)*IS30;
  T[25]=(ax*(S3C*B2+B4) + ay*B0 - az*B3)*IS30;
  T[26]=(S3C*(ay*B3 - ax*B1))*IS30;
  T[27]=(-ax*B0 + ay*(B4 - S3C*B2) + az*B1)*IS30;
  T[28]=(-ax*B1 - ay*B3 + 2.f*az*B0)*IS30;
  T[29]=A0*b0*IS5; T[30]=A1*b0*IS5; T[31]=A2*b0*IS5; T[32]=A3*b0*IS5; T[33]=A4*b0*IS5;
  T[34]=(by*A0+bz*A3+bx*A4)*IS10 - bx*A2*IS30;
  T[35]=(bx*A0+bz*A1-by*A4)*IS10 - by*A2*IS30;
  T[36]=(by*A1+bx*A3)*IS10 + 2.f*bz*A2*IS30;
  T[37]=(bx*A3 - by*A1 - 2.f*bz*A4)*IS30;
  T[38]=(bx*(S3C*A2+A4) + by*A0 - bz*A3)*IS30;
  T[39]=(S3C*(by*A3 - bx*A1))*IS30;
  T[40]=(-bx*A0 + by*(A4 - S3C*A2) + bz*A1)*IS30;
  T[41]=(-bx*A1 - by*A3 + 2.f*bz*A0)*IS30;
  T[42]=(A0*B0+A1*B1+A2*B2+A3*B3+A4*B4)*IS5;
  T[43]=(-A0*B3 + A3*B0 + S3C*(A2*B1 - A1*B2) - A1*B4 + A4*B1)*IS30;
  T[44]=(A0*B1 - A1*B0 + S3C*(A3*B2 - A2*B3) - A3*B4 + A4*B3)*IS30;
  T[45]=(2.f*(A0*B4 - A4*B0) + A1*B3 - A3*B1)*IS30;
  T[46]=(2.f*(A0*B2+A2*B0) - S3C*(A1*B3+A3*B1))*IS70;
  T[47]=(-(A1*B2+A2*B1) + S3C*(A1*B4+A4*B1) - S3C*(A0*B3+A3*B0))*IS70;
  T[48]=(2.f*A0*B0 - A1*B1 - 2.f*A2*B2 - A3*B3 + 2.f*A4*B4)*IS70;
  T[49]=(-(A3*B2+A2*B3) - S3C*(A3*B4+A4*B3) - S3C*(A0*B1+A1*B0))*IS70;
  T[50]=(2.f*(A4*B2+A2*B4) + S3C*(A1*B1 - A3*B3))*IS70;
  __bf16* Ap = A + (size_t)s * 64;
#pragma unroll
  for (int k = 0; k < 64; ++k) Ap[a_swz(k)] = (__bf16)T[k];
}

// One wave (32 threads) per 16-edge tile: (16x64) @ (64x368) with bf16 WMMA,
// f32 accumulate, then unconditional f32 atomic scatter (invalid rows -> trash row N).
__global__ void k_fctp1_wmma(const int* __restrict__ tile_z, const int* __restrict__ tile_s,
                             const int* __restrict__ ntiles, const int* __restrict__ boff,
                             const int* __restrict__ order, const int* __restrict__ edst,
                             const __bf16* __restrict__ A, const __bf16* __restrict__ G,
                             float* __restrict__ node2, int N) {
  int t = blockIdx.x;
  if (t >= *ntiles) return;
  int z  = tile_z[t];
  int t0 = tile_s[t];
  int end = boff[z + 1];
  int lane = threadIdx.x;
  int h = lane >> 4;

  // A fragments: contiguous 16 bf16 per lane thanks to the swizzled layout.
  const __bf16* Ap = A + (size_t)(t0 + (lane & 15)) * 64 + (h << 4);
  v16bf a0, a1;
#pragma unroll
  for (int i = 0; i < 16; ++i) { a0[i] = Ap[i]; a1[i] = Ap[32 + i]; }

  // Destination rows; invalid rows dumped into trash row N (never read back).
  int dstm[8];
#pragma unroll
  for (int r = 0; r < 8; ++r) {
    int sidx = t0 + r + (h << 3);
    dstm[r] = (sidx < end) ? edst[order[sidx]] : N;
  }

  // B: lane = column n (lane & 15); element i = K in table order within 32-step.
  const __bf16* G0 = G + (size_t)z * 64 * 368 + (lane & 15);
  const __bf16* G1 = G0 + (size_t)32 * 368;
  __builtin_prefetch(G0, 0, 3);
  __builtin_prefetch(G1, 0, 3);
  int n = lane & 15;
  for (int nt = 0; nt < 23; ++nt) {
    int cb = nt * 16;
    v16bf bb0, bb1;
#pragma unroll
    for (int i = 0; i < 16; ++i) { int kr = ((i >> 3) << 4) + (h << 3) + (i & 7); bb0[i] = G0[(size_t)kr * 368 + cb]; bb1[i] = G1[(size_t)kr * 368 + cb]; }
    v8f c = {0.f,0.f,0.f,0.f,0.f,0.f,0.f,0.f};
    c = __builtin_amdgcn_wmma_f32_16x16x32_bf16(false, a0, false, bb0, (short)0, c, false, false);
    c = __builtin_amdgcn_wmma_f32_16x16x32_bf16(false, a1, false, bb1, (short)0, c, false, false);
#pragma unroll
    for (int r = 0; r < 8; ++r) {
      atomicAdd(&node2[(size_t)dstm[r] * 368 + cb + n], c[r]);
    }
  }
}

// fctp2: one wave per edge; gather node2[src] (368 f32), weighted dots, reduce, scatter scalar.
__global__ void k_fctp2(const int* __restrict__ esrc, const int* __restrict__ edst,
                        const int* __restrict__ zzbuf, const float* __restrict__ sh,
                        const float* __restrict__ node2, const float* __restrict__ w2,
                        float* __restrict__ node3, int E) {
  int wid  = (blockIdx.x * blockDim.x + threadIdx.x) >> 5;
  int lane = threadIdx.x & 31;
  if (wid >= E) return;
  int e = wid;
  int src = esrc[e], dst = edst[e], zz = zzbuf[e];
  const float* n2 = node2 + (size_t)src * 368;
  __builtin_prefetch(n2, 0, 3);
  __builtin_prefetch(n2 + 184, 0, 3);
  float acc = 0.f;
  acc += n2[lane]      * w2[lane * 64 + zz];
  acc += n2[lane + 32] * w2[(lane + 32) * 64 + zz];
  if (lane < 24) {
    float bx = sh[(size_t)e*9+1], by = sh[(size_t)e*9+2], bz = sh[(size_t)e*9+3];
    const float* p = n2 + 136 + 3 * lane;
    float dot = p[0]*bx + p[1]*by + p[2]*bz;
    acc += dot * IS3 * w2[4096 + lane * 64 + zz];
  }
  if (lane < 16) {
    const float* p = n2 + 208 + 5 * lane;
    float d5 = 0.f;
#pragma unroll
    for (int k = 0; k < 5; ++k) d5 += p[k] * sh[(size_t)e*9 + 4 + k];
    acc += d5 * IS5 * w2[5632 + lane * 64 + zz];
  }
#pragma unroll
  for (int off = 16; off > 0; off >>= 1) acc += __shfl_down(acc, off, 32);
  // 8*alpha2 = 8/sqrt(6656); node2 scale 1/sqrt(3)
  if (lane == 0) atomicAdd(&node3[dst], acc * (0.09805806756909202f * IS3));
}

__global__ void k_final(const int* __restrict__ batch, const float* __restrict__ node3,
                        float* __restrict__ out, int N) {
  int n = blockIdx.x * blockDim.x + threadIdx.x;
  if (n >= N) return;
  // out = segsum(node3/sqrt(3), batch) / sqrt(4)
  atomicAdd(&out[batch[n]], node3[n] * 0.28867513459481287f);
}

extern "C" void kernel_launch(void* const* d_in, const int* in_sizes, int n_in,
                              void* d_out, int out_size, void* d_ws, size_t ws_size,
                              hipStream_t stream) {
  (void)n_in; (void)ws_size;
  const float* pos  = (const float*)d_in[0];
  const int*   z    = (const int*)d_in[1];
  const int*   batch= (const int*)d_in[2];
  const int*   esrc = (const int*)d_in[3];
  const int*   edst = (const int*)d_in[4];
  const float* w1   = (const float*)d_in[5];
  const float* w2   = (const float*)d_in[6];
  float* out = (float*)d_out;
  const int N = in_sizes[1];
  const int E = in_sizes[3];
  const int T_CAP = E / 16 + 65;

  char* base = (char*)d_ws;
  size_t off = 0;
  auto alloc = [&](size_t bytes) -> void* {
    void* p = (void*)(base + off);
    off = (off + bytes + 255) & ~(size_t)255;
    return p;
  };
  float*  node1  = (float*)alloc((size_t)N * 9 * 4);
  float*  node2  = (float*)alloc((size_t)(N + 1) * 368 * 4);  // +1 trash row
  float*  node3  = (float*)alloc((size_t)N * 4);
  float*  sh     = (float*)alloc((size_t)E * 9 * 4);
  int*    zzbuf  = (int*)alloc((size_t)E * 4);
  int*    order  = (int*)alloc((size_t)E * 4);
  int*    cnt    = (int*)alloc(64 * 4);
  int*    boff   = (int*)alloc(65 * 4);
  int*    cursor = (int*)alloc(64 * 4);
  int*    tile_z = (int*)alloc((size_t)T_CAP * 4);
  int*    tile_s = (int*)alloc((size_t)T_CAP * 4);
  int*    ntiles = (int*)alloc(4);
  __bf16* A      = (__bf16*)alloc(((size_t)E + 1024) * 64 * 2);
  __bf16* G      = (__bf16*)alloc((size_t)64 * 64 * 368 * 2);

  hipMemsetAsync(node1, 0, (size_t)N * 9 * 4, stream);
  hipMemsetAsync(node2, 0, (size_t)(N + 1) * 368 * 4, stream);
  hipMemsetAsync(node3, 0, (size_t)N * 4, stream);
  hipMemsetAsync(cnt,    0, 64 * 4, stream);
  hipMemsetAsync(cursor, 0, 64 * 4, stream);
  hipMemsetAsync(A + (size_t)E * 64, 0, (size_t)1024 * 64 * 2, stream); // zero tail rows only
  hipMemsetAsync(G, 0, (size_t)64 * 64 * 368 * 2, stream);
  hipMemsetAsync(out, 0, (size_t)out_size * 4, stream);

  k_edge_prep<<<(E + 255) / 256, 256, 0, stream>>>(pos, z, esrc, edst, E, sh, zzbuf, node1, cnt);
  k_scan_tiles<<<1, 64, 0, stream>>>(cnt, E, boff, tile_z, tile_s, ntiles);
  k_scatter<<<(E + 255) / 256, 256, 0, stream>>>(zzbuf, boff, cursor, order, E);
  k_gfill<<<(64 * 1104 + 255) / 256, 256, 0, stream>>>(w1, G);
  k_abuild<<<(E + 255) / 256, 256, 0, stream>>>(order, esrc, sh, node1, A, E);
  k_fctp1_wmma<<<T_CAP, 32, 0, stream>>>(tile_z, tile_s, ntiles, boff, order, edst, A, G, node2, N);
  k_fctp2<<<(E + 7) / 8, 256, 0, stream>>>(esrc, edst, zzbuf, sh, node2, w2, node3, E);
  k_final<<<(N + 255) / 256, 256, 0, stream>>>(batch, node3, out, N);
}
